// Generator_70935679861225
// MI455X (gfx1250) — matrix operand order repaired and checked by reading, not compiled, not run on hardware
//
#include <hip/hip_runtime.h>
#include <hip/hip_bf16.h>
#include <stdint.h>

// Problem constants
#define V_  32000
#define E_  300
#define H_  300
#define RD_ 50
#define B_  32
#define N_  128
#define G4_ 1200          // 4*H
#define IN_ 350           // RD + E
#define KP_ 320           // K padded to multiple of 32 for wmma (300 -> 320)
#define KB_ 10            // KP_/32 k-blocks
#define MT_ 256           // (B*N)/16 m-tiles
#define NT_ 2000          // V/16 n-tiles
#define FRAG 512          // elems per packed (tile,kblock) fragment: 32 lanes * 16 bf16

typedef __attribute__((ext_vector_type(16))) __bf16 v16bf;
typedef __attribute__((ext_vector_type(8)))  float  v8f;

// ---------- helpers ----------
__device__ inline unsigned short f32_to_bf16_rne(float f) {
    unsigned int u = __float_as_uint(f);
    unsigned int lsb = (u >> 16) & 1u;
    u += 0x7fffu + lsb;
    return (unsigned short)(u >> 16);
}
__device__ inline float sigmoidf_(float x) { return 1.0f / (1.0f + expf(-x)); }

// ---------- 1) transpose W_ih [1200,350] -> W_ih_t [350,1200] ----------
__global__ void k_transpose_wih(const float* __restrict__ W, float* __restrict__ Wt) {
    for (int idx = blockIdx.x * blockDim.x + threadIdx.x; idx < G4_ * IN_;
         idx += gridDim.x * blockDim.x) {
        int g = idx / IN_, i = idx % IN_;
        Wt[i * G4_ + g] = W[idx];
    }
}

// ---------- 2) transpose W_hh [1200,300] -> W_hh_t [300,1200] ----------
__global__ void k_transpose_whh(const float* __restrict__ W, float* __restrict__ Wt) {
    for (int idx = blockIdx.x * blockDim.x + threadIdx.x; idx < G4_ * H_;
         idx += gridDim.x * blockDim.x) {
        int g = idx / H_, k = idx % H_;
        Wt[k * G4_ + g] = W[idx];
    }
}

// ---------- 3) pack W_out [32000,300] f32 -> bf16 B-fragments [NT][KB][32][16] ----------
// B matrix (KxN, 16-bit) layout used here (contiguous K, not the ISA table order): lanes 0-15 hold K=0..15 (N=lane),
// lanes 16-31 hold K=16..31 (N=lane-16); element e within lane = K offset e.
__global__ void k_pack_wout(const float* __restrict__ Wout, unsigned short* __restrict__ Bpk) {
    for (long long idx = blockIdx.x * (long long)blockDim.x + threadIdx.x;
         idx < (long long)NT_ * KB_ * FRAG; idx += (long long)gridDim.x * blockDim.x) {
        long long t0 = idx;
        int e    = (int)(t0 & 15); t0 >>= 4;
        int lane = (int)(t0 & 31); t0 >>= 5;
        int kb   = (int)(t0 % KB_);
        int t    = (int)(t0 / KB_);
        int k = kb * 32 + e + 16 * (lane >> 4);
        int n = t * 16 + (lane & 15);
        float v = (k < H_) ? Wout[(size_t)n * H_ + k] : 0.0f;
        Bpk[idx] = f32_to_bf16_rne(v);
    }
}

// ---------- 4) xg = x . W_ih^T + b_ih ; x = rel_emb[rel] ++ emb[word] ----------
#define XG_MB 4
__global__ __launch_bounds__(256) void k_xg(const int* __restrict__ relations,
                                            const int* __restrict__ prev_words,
                                            const float* __restrict__ emb,
                                            const float* __restrict__ rel_emb,
                                            const float* __restrict__ Wih_t,
                                            const float* __restrict__ b_ih,
                                            float* __restrict__ xg) {
    __shared__ float xs[XG_MB * IN_];
    int m0 = blockIdx.x * XG_MB;
    // stage the XG_MB gathered input vectors in LDS
    for (int idx = threadIdx.x; idx < XG_MB * IN_; idx += blockDim.x) {
        int ml = idx / IN_, i = idx % IN_;
        int m = m0 + ml;
        float v;
        if (i < RD_) v = rel_emb[relations[m] * RD_ + i];
        else         v = emb[(size_t)prev_words[m] * E_ + (i - RD_)];
        xs[idx] = v;
    }
    __syncthreads();
    for (int g = threadIdx.x; g < G4_; g += blockDim.x) {
        float a0 = 0.f, a1 = 0.f, a2 = 0.f, a3 = 0.f;
        for (int i = 0; i < IN_; ++i) {
            float w = Wih_t[i * G4_ + g];           // coalesced across g
            a0 = fmaf(w, xs[i], a0);
            a1 = fmaf(w, xs[IN_ + i], a1);
            a2 = fmaf(w, xs[2 * IN_ + i], a2);
            a3 = fmaf(w, xs[3 * IN_ + i], a3);
        }
        float bi = b_ih[g];
        xg[(size_t)(m0 + 0) * G4_ + g] = a0 + bi;
        xg[(size_t)(m0 + 1) * G4_ + g] = a1 + bi;
        xg[(size_t)(m0 + 2) * G4_ + g] = a2 + bi;
        xg[(size_t)(m0 + 3) * G4_ + g] = a3 + bi;
    }
}

// ---------- 5) tree LSTM: one workgroup per tree, (N+1)*H h/c buffers in LDS ----------
// Dynamic LDS: 2 * 129 * 300 floats = 309,600 B (needs CDNA5's 320 KB WGP LDS).
__global__ __launch_bounds__(320) void k_tree_lstm(const float* __restrict__ xg,
                                                   const float* __restrict__ root_h,
                                                   const int* __restrict__ parents,
                                                   const float* __restrict__ Whh_t,
                                                   const float* __restrict__ b_hh,
                                                   float* __restrict__ hs) {
    extern __shared__ float smem[];
    float* hb = smem;                 // [(N+1)*H]
    float* cb = smem + (N_ + 1) * H_; // [(N+1)*H]
    int b = blockIdx.x;
    for (int j = threadIdx.x; j < H_; j += blockDim.x) {
        float r = root_h[b * H_ + j];
        hb[j] = r; cb[j] = r;         // reference seeds (h, c) = (h0, h0)
    }
    __syncthreads();
    for (int i = 0; i < N_; ++i) {
        int par = parents[b * N_ + i];
        const float* ph = hb + (par + 1) * H_;
        const float* pc = cb + (par + 1) * H_;
        int u = threadIdx.x;
        if (u < H_) {
            const float* xgi = xg + (size_t)(b * N_ + i) * G4_;
            float gi = xgi[u]            + b_hh[u];
            float gf = xgi[H_ + u]       + b_hh[H_ + u];
            float gg = xgi[2 * H_ + u]   + b_hh[2 * H_ + u];
            float go = xgi[3 * H_ + u]   + b_hh[3 * H_ + u];
            for (int k = 0; k < H_; ++k) {
                float h = ph[k];
                const float* wr = Whh_t + k * G4_;   // coalesced across u
                gi = fmaf(wr[u], h, gi);
                gf = fmaf(wr[H_ + u], h, gf);
                gg = fmaf(wr[2 * H_ + u], h, gg);
                go = fmaf(wr[3 * H_ + u], h, go);
            }
            float c = sigmoidf_(gf) * pc[u] + sigmoidf_(gi) * tanhf(gg);
            float h = sigmoidf_(go) * tanhf(c);
            hb[(i + 1) * H_ + u] = h;
            cb[(i + 1) * H_ + u] = c;
            hs[(size_t)(b * N_ + i) * H_ + u] = h;
        }
        __syncthreads();
    }
}

// ---------- 6) pack hs [4096,300] f32 -> bf16 A-fragments [MT][KB][32][16] ----------
// A matrix (MxK, 16-bit) layout: M = lane&15; lanes<16: K = 0..15,
// lanes>=16: K = 16..31; element e -> K = e + 16*(lane>>4).
__global__ void k_pack_hs(const float* __restrict__ hs, unsigned short* __restrict__ Apk) {
    for (long long idx = blockIdx.x * (long long)blockDim.x + threadIdx.x;
         idx < (long long)MT_ * KB_ * FRAG; idx += (long long)gridDim.x * blockDim.x) {
        long long t0 = idx;
        int e    = (int)(t0 & 15); t0 >>= 4;
        int lane = (int)(t0 & 31); t0 >>= 5;
        int kb   = (int)(t0 % KB_);
        int t    = (int)(t0 / KB_);
        int k = kb * 32 + e + 16 * (lane >> 4);
        int m = t * 16 + (lane & 15);
        float v = (k < H_) ? hs[(size_t)m * H_ + k] : 0.0f;
        Apk[idx] = f32_to_bf16_rne(v);
    }
}

// ---------- 7) logits GEMM: [4096 x 32000], K=320, bf16 WMMA 16x16x32 ----------
// Block = 256 threads = 8 waves (wave32), arranged 2(Mgrp) x 4(N).
// Wave tile: 32(M) x 64(N) = 8 accumulators; B fragments reused for both m-tiles
// (8 WMMAs per 6 fragment loads). Block tile: 64 x 256.
// Grid: (32000/256, 4096/64) = (125, 64).
__global__ __launch_bounds__(256, 1) void k_gemm_out(const unsigned short* __restrict__ Apk,
                                                     const unsigned short* __restrict__ Bpk,
                                                     const float* __restrict__ b_out,
                                                     float* __restrict__ out) {
    int lane = threadIdx.x & 31;
    int wave = threadIdx.x >> 5;
    int wm = wave >> 2;            // 0..1
    int wn = wave & 3;             // 0..3
    int mt0 = blockIdx.y * 4 + wm * 2;         // first of 2 m-tiles for this wave
    int nt  = blockIdx.x * 16 + wn * 4;        // first of 4 n-tiles for this wave

    const unsigned short* aBase = Apk + (size_t)mt0 * KB_ * FRAG + lane * 16;
    const unsigned short* bBase = Bpk + (size_t)nt * KB_ * FRAG + lane * 16;

    v8f acc[2][4] = {};
#pragma unroll
    for (int kb = 0; kb < KB_; ++kb) {
        v16bf a0 = *(const v16bf*)(aBase + kb * FRAG);
        v16bf a1 = *(const v16bf*)(aBase + (KB_ + kb) * FRAG);
        v16bf b0 = *(const v16bf*)(bBase + kb * FRAG);
        v16bf b1 = *(const v16bf*)(bBase + (1 * KB_ + kb) * FRAG);
        v16bf b2 = *(const v16bf*)(bBase + (2 * KB_ + kb) * FRAG);
        v16bf b3 = *(const v16bf*)(bBase + (3 * KB_ + kb) * FRAG);
        if (kb + 1 < KB_) {  // hint next K-block toward the WGP (global_prefetch_b8)
            __builtin_prefetch(aBase + (kb + 1) * FRAG, 0, 1);
            __builtin_prefetch(bBase + (kb + 1) * FRAG, 0, 1);
        }
        acc[0][0] = __builtin_amdgcn_wmma_f32_16x16x32_bf16(false, a0, false, b0, (short)0, acc[0][0], false, false);
        acc[0][1] = __builtin_amdgcn_wmma_f32_16x16x32_bf16(false, a0, false, b1, (short)0, acc[0][1], false, false);
        acc[0][2] = __builtin_amdgcn_wmma_f32_16x16x32_bf16(false, a0, false, b2, (short)0, acc[0][2], false, false);
        acc[0][3] = __builtin_amdgcn_wmma_f32_16x16x32_bf16(false, a0, false, b3, (short)0, acc[0][3], false, false);
        acc[1][0] = __builtin_amdgcn_wmma_f32_16x16x32_bf16(false, a1, false, b0, (short)0, acc[1][0], false, false);
        acc[1][1] = __builtin_amdgcn_wmma_f32_16x16x32_bf16(false, a1, false, b1, (short)0, acc[1][1], false, false);
        acc[1][2] = __builtin_amdgcn_wmma_f32_16x16x32_bf16(false, a1, false, b2, (short)0, acc[1][2], false, false);
        acc[1][3] = __builtin_amdgcn_wmma_f32_16x16x32_bf16(false, a1, false, b3, (short)0, acc[1][3], false, false);
    }
    // C/D layout: VGPR j, lane l -> (M = j + 8*(l>>4), N = l&15)
    int mrow = mt0 * 16 + 8 * (lane >> 4);
    int n0   = nt * 16 + (lane & 15);
    float bo0 = b_out[n0], bo1 = b_out[n0 + 16], bo2 = b_out[n0 + 32], bo3 = b_out[n0 + 48];
#pragma unroll
    for (int mi = 0; mi < 2; ++mi) {
#pragma unroll
        for (int j = 0; j < 8; ++j) {
            // streaming output: non-temporal stores keep W_out fragments resident in L2
            float* orow = out + (size_t)(mrow + mi * 16 + j) * V_;
            __builtin_nontemporal_store(acc[mi][0][j] + bo0, orow + n0);
            __builtin_nontemporal_store(acc[mi][1][j] + bo1, orow + n0 + 16);
            __builtin_nontemporal_store(acc[mi][2][j] + bo2, orow + n0 + 32);
            __builtin_nontemporal_store(acc[mi][3][j] + bo3, orow + n0 + 48);
        }
    }
}

// ---------- launcher ----------
extern "C" void kernel_launch(void* const* d_in, const int* in_sizes, int n_in,
                              void* d_out, int out_size, void* d_ws, size_t ws_size,
                              hipStream_t stream) {
    const float* root_hidden = (const float*)d_in[0];
    const int*   relations   = (const int*)d_in[1];
    const int*   prev_words  = (const int*)d_in[2];
    const int*   parents     = (const int*)d_in[3];
    const float* emb         = (const float*)d_in[4];
    const float* rel_emb     = (const float*)d_in[5];
    const float* W_ih        = (const float*)d_in[6];
    const float* W_hh        = (const float*)d_in[7];
    const float* b_ih        = (const float*)d_in[8];
    const float* b_hh        = (const float*)d_in[9];
    const float* W_out       = (const float*)d_in[10];
    const float* b_out       = (const float*)d_in[11];
    float* out = (float*)d_out;

    // Workspace layout (all offsets 32B-aligned; total ~50.8 MB)
    char* ws = (char*)d_ws;
    float* xg      = (float*)(ws);                                    // 4096*1200*4 = 19,660,800
    float* hs      = (float*)(ws + 19660800);                         //  4096*300*4 =  4,915,200
    float* Whh_t   = (float*)(ws + 24576000);                         //  300*1200*4 =  1,440,000
    float* Wih_t   = (float*)(ws + 26016000);                         //  350*1200*4 =  1,680,000
    unsigned short* Apk = (unsigned short*)(ws + 27696000);           //  256*10*512*2 =  2,621,440
    unsigned short* Bpk = (unsigned short*)(ws + 30317440);           // 2000*10*512*2 = 20,480,000
    (void)ws_size; (void)in_sizes; (void)n_in; (void)out_size;

    // Stage 0: weight transposes + W_out bf16 fragment pack (independent)
    k_transpose_wih<<<512, 256, 0, stream>>>(W_ih, Wih_t);
    k_transpose_whh<<<512, 256, 0, stream>>>(W_hh, Whh_t);
    k_pack_wout<<<4096, 256, 0, stream>>>(W_out, Bpk);

    // Stage 1: per-node gate pre-activations
    k_xg<<<(B_ * N_) / XG_MB, 256, 0, stream>>>(relations, prev_words, emb, rel_emb,
                                                Wih_t, b_ih, xg);

    // Stage 2: serial tree recurrence, one tree per WGP, h/c in 302 KB LDS
    size_t lds = (size_t)2 * (N_ + 1) * H_ * sizeof(float);
    k_tree_lstm<<<B_, 320, lds, stream>>>(xg, root_hidden, parents, Whh_t, b_hh, hs);

    // Stage 3: pack hs into bf16 A-fragments
    k_pack_hs<<<1024, 256, 0, stream>>>(hs, Apk);

    // Stage 4: logits = hs @ W_out^T + b_out via v_wmma_f32_16x16x32_bf16
    dim3 grid(V_ / 256, (B_ * N_) / 64);
    k_gemm_out<<<grid, 256, 0, stream>>>(Apk, Bpk, b_out, out);
}
